// DeepKalmanFilter_84808424227118
// MI455X (gfx1250) — compile-verified, matrix operand order checked by reading, not run on hardware
//
#include <hip/hip_runtime.h>
#include <hip/hip_bf16.h>

typedef __attribute__((ext_vector_type(16))) _Float16 v16h;
typedef __attribute__((ext_vector_type(8)))  float    v8f;

#define T_LEN 256
#define B_N   64
#define H_N   256
#define D_IN  128
#define S_N   32
#define ST_N  64
#define G4H   1024   // 4*H
#define K_TOT 384    // D_IN + H_N
#define FRAG_H 512   // halves per packed B fragment (32 lanes x 16)

// ---------------------------------------------------------------------------
// WMMA fragment helpers (ISA 7.12.2 layouts, wave32)
// ---------------------------------------------------------------------------
__device__ __forceinline__ int lane_id() { return threadIdx.x & 31; }

// A-matrix 16x32 f16, row-major source with leading dim lda.
__device__ __forceinline__ v16h load_frag_a(const _Float16* a, int lda) {
  const int l  = lane_id();
  const int r  = l & 15;
  const int kh = (l >> 4) << 3;
  const _Float16* p = a + r * lda + kh;
  v16h v;
#pragma unroll
  for (int i = 0; i < 8; ++i) { v[i] = p[i]; v[i + 8] = p[16 + i]; }
  return v;
}

// B fragment from packed (fragment-major) storage: one contiguous 32B per lane.
__device__ __forceinline__ v16h load_frag_b_packed(const _Float16* frag) {
  return *(const v16h*)(frag + lane_id() * 16);
}

// fast activations built on v_exp2 / v_rcp (saturate correctly at +/- inf)
__device__ __forceinline__ float fast_sigmoid(float x) {
  const float t = __builtin_amdgcn_exp2f(x * -1.442695040888963f);
  return __builtin_amdgcn_rcpf(1.0f + t);
}
__device__ __forceinline__ float fast_tanh(float x) {
  const float t = __builtin_amdgcn_exp2f(x * 2.885390081777927f);
  return 1.0f - 2.0f * __builtin_amdgcn_rcpf(t + 1.0f);
}

// ---------------------------------------------------------------------------
// Staging kernels
// ---------------------------------------------------------------------------
// y [B,T,D] fp32 -> yh [T,B,D] f16
__global__ void transpose_y_kernel(const float* __restrict__ y,
                                   _Float16* __restrict__ yh) {
  const int n = T_LEN * B_N * D_IN;
  for (int i = blockIdx.x * blockDim.x + threadIdx.x; i < n;
       i += gridDim.x * blockDim.x) {
    const int d = i % D_IN;
    const int b = (i / D_IN) % B_N;
    const int t = i / (D_IN * B_N);
    yh[i] = (_Float16)y[((size_t)b * T_LEN + t) * D_IN + d];
  }
}

// Pack a row-major fp32 weight [Kloc, N] into f16 fragment-major B layout.
__global__ void pack_b_kernel(const float* __restrict__ src,
                              _Float16* __restrict__ dst,
                              int Kloc, int N, int kbOff) {
  const int total = Kloc * N;
  const int NT = N >> 4;
  for (int e = blockIdx.x * blockDim.x + threadIdx.x; e < total;
       e += gridDim.x * blockDim.x) {
    const int fl   = e >> 9;          // local fragment index
    const int w    = e & 511;
    const int lane = w >> 4;
    const int i    = w & 15;
    const int kb   = fl / NT;
    const int nt   = fl % NT;
    const int n    = nt * 16 + (lane & 15);
    const int kh   = (lane >> 4) << 3;
    const int kk   = (i < 8) ? (kh + i) : (kh + 16 + (i - 8));
    const int k    = kb * 32 + kk;
    dst[((size_t)(kbOff + kb) * NT + nt) * FRAG_H + lane * 16 + i] =
        (_Float16)src[(size_t)k * N + n];
  }
}

// ---------------------------------------------------------------------------
// Bidirectional LSTM scan. grid = 8 blocks (dir x 4 batch tiles of M=16),
// block = 512 threads = 16 waves (4 N-tiles of the 1024 gate cols each).
// B weights re-stream from L2 every step; a laundered zero OFFSET (not a
// laundered pointer) defeats LICM while keeping the global address space,
// so weight traffic lowers to global_load_b128 (LOADcnt only, no flat).
// ---------------------------------------------------------------------------
__global__ void lstm_scan_kernel(const _Float16* __restrict__ yh,
                                 const _Float16* __restrict__ Wpkf,
                                 const _Float16* __restrict__ Wpkb,
                                 const float* __restrict__ bf,
                                 const float* __restrict__ bb,
                                 float* __restrict__ hidf,
                                 float* __restrict__ hidb) {
  extern __shared__ __align__(32) char smem[];
  _Float16* xL = (_Float16*)smem;          // 16 x 128 f16  (4 KB)
  _Float16* hL = xL + 16 * D_IN;           // 16 x 256 f16  (8 KB)
  float*    cL = (float*)(hL + 16 * H_N);  // 16 x 256 f32  (16 KB)
  float*    zL = cL + 16 * H_N;            // 16 x 1024 f32 (64 KB)

  const int dir = blockIdx.x >> 2;               // 0 = fwd, 1 = bwd
  const int b0  = (blockIdx.x & 3) * 16;
  const _Float16* Wpk  = dir ? Wpkb : Wpkf;      // packed fragments
  const float*    bias = dir ? bb : bf;          // [1024] fp32
  float*          hid  = dir ? hidb : hidf;      // [T,B,H] fp32

  const int tid = threadIdx.x;
  const int wv  = tid >> 5;                      // wave id 0..15

  for (int i = tid; i < 16 * H_N; i += blockDim.x) {
    cL[i] = 0.0f;
    hL[i] = (_Float16)0.0f;
  }
  __syncthreads();

  for (int step = 0; step < T_LEN; ++step) {
    const int t = dir ? (T_LEN - 1 - step) : step;

    // opaque zero offset, renewed every step: the 48 B-fragment loads can't
    // be proven loop-invariant, so they are not hoisted and spilled, yet the
    // base pointer keeps its global address space.
    size_t zoff = 0;
    asm volatile("" : "+s"(zoff));
    const _Float16* Wstep = Wpk + zoff;

    // stage x_t tile (16 batch rows are contiguous in yh[t])
    const _Float16* xsrc = yh + ((size_t)t * B_N + b0) * D_IN;
    for (int i = tid; i < 16 * D_IN; i += blockDim.x) xL[i] = xsrc[i];
    __syncthreads();

    // GEMM: 4 accumulators, K loop of 12 x 32, double-buffered B fragments
    v8f acc[4];
#pragma unroll
    for (int j = 0; j < 4; ++j)
#pragma unroll
      for (int e = 0; e < 8; ++e) acc[j][e] = 0.0f;

    v16h bcur[4], bnxt[4];
    {
      const _Float16* fb = Wstep + (size_t)(wv * 4) * FRAG_H;
#pragma unroll
      for (int j = 0; j < 4; ++j) bcur[j] = load_frag_b_packed(fb + (size_t)j * FRAG_H);
    }
#pragma unroll
    for (int kb = 0; kb < 12; ++kb) {
      if (kb + 1 < 12) {
        const _Float16* fb = Wstep + ((size_t)(kb + 1) * 64 + wv * 4) * FRAG_H;
#pragma unroll
        for (int j = 0; j < 4; ++j) bnxt[j] = load_frag_b_packed(fb + (size_t)j * FRAG_H);
      }
      if (kb + 2 < 12)
        __builtin_prefetch(Wstep + ((size_t)(kb + 2) * 64 + wv * 4) * FRAG_H, 0, 1);
      const v16h afr = (kb < 4) ? load_frag_a(xL + kb * 32, D_IN)
                                : load_frag_a(hL + (kb - 4) * 32, H_N);
#pragma unroll
      for (int j = 0; j < 4; ++j)
        acc[j] = __builtin_amdgcn_wmma_f32_16x16x32_f16(
            false, afr, false, bcur[j], (short)0, acc[j], false, false);
#pragma unroll
      for (int j = 0; j < 4; ++j) bcur[j] = bnxt[j];
    }

    // spill D-fragments to zL per the C/D layout (lane = col, 8 rows/lane)
    {
      const int l    = lane_id();
      const int col0 = (wv * 4) * 16 + (l & 15);
      const int mh   = (l >> 4) * 8;
#pragma unroll
      for (int j = 0; j < 4; ++j)
#pragma unroll
        for (int e = 0; e < 8; ++e)
          zL[(mh + e) * G4H + col0 + j * 16] = acc[j][e];
    }
    __syncthreads();

    // fused gate activation: i,f,g,o laid out [0,256)[256,512)[512,768)[768,1024)
    for (int e = tid; e < 16 * H_N; e += blockDim.x) {
      const int r  = e >> 8;
      const int hc = e & 255;
      const float zi = zL[r * G4H + hc]       + bias[hc];
      const float zf = zL[r * G4H + 256 + hc] + bias[256 + hc];
      const float zg = zL[r * G4H + 512 + hc] + bias[512 + hc];
      const float zo = zL[r * G4H + 768 + hc] + bias[768 + hc];
      const float c  = fast_sigmoid(zf) * cL[e] + fast_sigmoid(zi) * fast_tanh(zg);
      const float h  = fast_sigmoid(zo) * fast_tanh(c);
      cL[e] = c;
      hL[e] = (_Float16)h;
      hid[((size_t)t * B_N + b0 + r) * H_N + hc] = h;
    }
    __syncthreads();
  }
}

// ---------------------------------------------------------------------------
// Sampling scan. grid = 128 blocks (s = blk>>2, b0 = (blk&3)*16),
// block = 256 threads = 8 waves. Each wave's 12 weight fragments are loop-
// invariant; they are staged in LDS once and re-read from LDS on each of the 256 steps.
// ---------------------------------------------------------------------------
__global__ void sample_scan_kernel(const float* __restrict__ hidf,
                                   const float* __restrict__ hidb,
                                   const _Float16* __restrict__ WtPk,   // 32 frags
                                   const _Float16* __restrict__ WmuPk,  // 32 frags
                                   const _Float16* __restrict__ WsgPk,  // 32 frags
                                   const float* __restrict__ bt,
                                   const float* __restrict__ bmu,
                                   const float* __restrict__ bsg,
                                   const float* __restrict__ eps,
                                   float* __restrict__ out) {
  extern __shared__ __align__(32) char smem[];
  _Float16* WtL  = (_Float16*)smem;          // 32 frags (32 KB)
  _Float16* WmuL = WtL + 32 * FRAG_H;        // 32 frags (32 KB)
  _Float16* WsgL = WmuL + 32 * FRAG_H;       // 32 frags (32 KB)
  _Float16* pL   = WsgL + 32 * FRAG_H;       // 16x64   (2 KB)
  _Float16* hcL  = pL + 16 * ST_N;           // 16x256  (8 KB)
  float*    muL  = (float*)(hcL + 16 * H_N); // 16x64 f32 (4 KB)
  float*    sgL  = muL + 16 * ST_N;          // 16x64 f32 (4 KB)

  const int tid  = threadIdx.x;
  const int wv   = tid >> 5;                 // wave id 0..7
  const int sIdx = blockIdx.x >> 2;          // sample index
  const int b0   = (blockIdx.x & 3) * 16;    // batch tile origin

  for (int i = tid; i < 32 * FRAG_H; i += blockDim.x) {
    WtL[i]  = WtPk[i];
    WmuL[i] = WmuPk[i];
    WsgL[i] = WsgPk[i];
  }
  for (int i = tid; i < 16 * ST_N; i += blockDim.x) pL[i] = (_Float16)0.0f;
  __syncthreads();

  const float inv3 = 1.0f / 3.0f;

  for (int t = 0; t < T_LEN; ++t) {
    // ---- GEMM1: [16,64] @ [64,256] -> 16 N-tiles over 8 waves ----
#pragma unroll
    for (int jt = 0; jt < 2; ++jt) {
      const int nt = wv * 2 + jt;
      v8f acc;
#pragma unroll
      for (int e = 0; e < 8; ++e) acc[e] = 0.0f;
#pragma unroll
      for (int kb = 0; kb < 2; ++kb) {
        const v16h a = load_frag_a(pL + kb * 32, ST_N);
        const v16h b = load_frag_b_packed(WtL + (size_t)(kb * 16 + nt) * FRAG_H);
        acc = __builtin_amdgcn_wmma_f32_16x16x32_f16(
            false, a, false, b, (short)0, acc, false, false);
      }
      const int l   = lane_id();
      const int col = nt * 16 + (l & 15);
      const int mh  = (l >> 4) * 8;
#pragma unroll
      for (int e = 0; e < 8; ++e) {
        const int r = mh + e;
        const size_t hoff = ((size_t)t * B_N + b0 + r) * H_N + col;
        const float th = fast_tanh(acc[e] + bt[col]);
        const float hc = inv3 * (th + hidf[hoff] + hidb[hoff]);
        hcL[r * H_N + col] = (_Float16)hc;
      }
    }
    __syncthreads();

    // ---- GEMM2/3: [16,256] @ [256,64]; waves 0-3 -> mu, 4-7 -> sg ----
    {
      const bool ismu     = (wv < 4);
      const int  nt       = ismu ? wv : (wv - 4);
      const _Float16* W   = ismu ? WmuL : WsgL;
      float*      stage   = ismu ? muL : sgL;
      const float* biasv  = ismu ? bmu : bsg;
      v8f acc;
#pragma unroll
      for (int e = 0; e < 8; ++e) acc[e] = 0.0f;
#pragma unroll
      for (int kb = 0; kb < 8; ++kb) {
        const v16h a = load_frag_a(hcL + kb * 32, H_N);
        const v16h b = load_frag_b_packed(W + (size_t)(kb * 4 + nt) * FRAG_H);
        acc = __builtin_amdgcn_wmma_f32_16x16x32_f16(
            false, a, false, b, (short)0, acc, false, false);
      }
      const int l   = lane_id();
      const int col = nt * 16 + (l & 15);
      const int mh  = (l >> 4) * 8;
#pragma unroll
      for (int e = 0; e < 8; ++e) stage[(mh + e) * ST_N + col] = acc[e] + biasv[col];
    }
    __syncthreads();

    // ---- reparameterized sample, write out + feed back into p ----
    for (int e = tid; e < 16 * ST_N; e += blockDim.x) {
      const int r = e >> 6;
      const int d = e & 63;
      const size_t eoff = (((size_t)t * S_N + sIdx) * B_N + (b0 + r)) * ST_N + d;
      const float s = muL[e] + sgL[e] * eps[eoff];
      out[(((size_t)sIdx * B_N + (b0 + r)) * T_LEN + t) * ST_N + d] = s;
      pL[e] = (_Float16)s;
    }
    __syncthreads();
  }
}

// ---------------------------------------------------------------------------
// launch
// ---------------------------------------------------------------------------
extern "C" void kernel_launch(void* const* d_in, const int* in_sizes, int n_in,
                              void* d_out, int out_size, void* d_ws, size_t ws_size,
                              hipStream_t stream) {
  const float* y    = (const float*)d_in[0];
  /* d_in[1] = n_samples (compile-time S_N) */
  const float* eps  = (const float*)d_in[2];
  const float* Wx_f = (const float*)d_in[3];
  const float* Wh_f = (const float*)d_in[4];
  const float* b_f  = (const float*)d_in[5];
  const float* Wx_b = (const float*)d_in[6];
  const float* Wh_b = (const float*)d_in[7];
  const float* b_b  = (const float*)d_in[8];
  const float* Wt   = (const float*)d_in[9];
  const float* bt   = (const float*)d_in[10];
  const float* Wmu  = (const float*)d_in[11];
  const float* bmu  = (const float*)d_in[12];
  const float* Wsg  = (const float*)d_in[13];
  const float* bsg  = (const float*)d_in[14];
  float* out = (float*)d_out;

  // workspace carve-up (~39.4 MB total)
  char* ws = (char*)d_ws;
  size_t off = 0;
  auto carve = [&](size_t bytes) {
    char* p = ws + off;
    off += (bytes + 255) & ~(size_t)255;
    return p;
  };
  _Float16* yh    = (_Float16*)carve((size_t)T_LEN * B_N * D_IN * 2);
  _Float16* Wpkf  = (_Float16*)carve((size_t)K_TOT * G4H * 2);   // 768 frags
  _Float16* Wpkb  = (_Float16*)carve((size_t)K_TOT * G4H * 2);
  _Float16* WtPk  = (_Float16*)carve((size_t)ST_N * H_N * 2);    // 32 frags
  _Float16* WmuPk = (_Float16*)carve((size_t)H_N * ST_N * 2);    // 32 frags
  _Float16* WsgPk = (_Float16*)carve((size_t)H_N * ST_N * 2);    // 32 frags
  float*    hidf  = (float*)carve((size_t)T_LEN * B_N * H_N * 4);
  float*    hidb  = (float*)carve((size_t)T_LEN * B_N * H_N * 4);

  constexpr int LSTM_SMEM = (16 * D_IN + 16 * H_N) * 2 + (16 * H_N + 16 * G4H) * 4; // 94208
  constexpr int SAMP_SMEM =
      (3 * 32 * FRAG_H + 16 * ST_N + 16 * H_N) * 2 + 2 * 16 * ST_N * 4;             // 116736
  (void)hipFuncSetAttribute((const void*)lstm_scan_kernel,
                            hipFuncAttributeMaxDynamicSharedMemorySize, LSTM_SMEM);
  (void)hipFuncSetAttribute((const void*)sample_scan_kernel,
                            hipFuncAttributeMaxDynamicSharedMemorySize, SAMP_SMEM);

  // stage f16 operands (weights go to fragment-major packed layout)
  transpose_y_kernel<<<2048, 256, 0, stream>>>(y, yh);
  pack_b_kernel<<<512, 256, 0, stream>>>(Wx_f, Wpkf, D_IN, G4H, 0);
  pack_b_kernel<<<1024, 256, 0, stream>>>(Wh_f, Wpkf, H_N, G4H, 4);
  pack_b_kernel<<<512, 256, 0, stream>>>(Wx_b, Wpkb, D_IN, G4H, 0);
  pack_b_kernel<<<1024, 256, 0, stream>>>(Wh_b, Wpkb, H_N, G4H, 4);
  pack_b_kernel<<<64, 256, 0, stream>>>(Wt, WtPk, ST_N, H_N, 0);
  pack_b_kernel<<<64, 256, 0, stream>>>(Wmu, WmuPk, H_N, ST_N, 0);
  pack_b_kernel<<<64, 256, 0, stream>>>(Wsg, WsgPk, H_N, ST_N, 0);

  // phase 1: bidirectional LSTM recurrence (2 dirs x 4 batch tiles)
  lstm_scan_kernel<<<8, 512, LSTM_SMEM, stream>>>(yh, Wpkf, Wpkb, b_f, b_b, hidf, hidb);

  // phase 2: sampling recurrence (128 independent (s,b) tiles)
  sample_scan_kernel<<<128, 256, SAMP_SMEM, stream>>>(hidf, hidb, WtPk, WmuPk, WsgPk,
                                                      bt, bmu, bsg, eps, out);
}
